// TransformerBlock_15049565405419
// MI455X (gfx1250) — matrix operand order repaired and checked by reading, compiled, not run on hardware
//
#include <hip/hip_runtime.h>
#include <hip/hip_bf16.h>
#include <cstdint>

// ---------------------------------------------------------------------------
// CDNA5 (gfx1250) wave32 WMMA transformer block with TDM tile staging.
//   D_MODEL=2048, N_HEADS=16, D_K=128, D_FF=5504, B=2, S=2048, M = B*S = 4096
// ---------------------------------------------------------------------------

typedef __attribute__((ext_vector_type(16))) __bf16    v16bf;
typedef __attribute__((ext_vector_type(8)))  float     v8f;
typedef __attribute__((ext_vector_type(4)))  unsigned  v4u;
typedef __attribute__((ext_vector_type(8)))  int       v8i;
typedef __attribute__((ext_vector_type(4)))  int       v4i;
typedef unsigned short u16;

union BF16Frag { u16 u[16]; uint4 q[2]; v16bf v; };

__device__ __forceinline__ u16 f2bf(float f) {
  unsigned u = __float_as_uint(f);
  unsigned r = u + 0x7FFFu + ((u >> 16) & 1u);   // round-to-nearest-even
  return (u16)(r >> 16);
}

// ---------------------------------------------------------------------------
// Tensor Data Mover: 2D tile (elements are 2 bytes) global -> LDS.
//   tile_x elements per row, tile_y rows, row stride (elements) in global.
//   D# packing per CDNA5 ISA 8.3/8.4 (group0: count/lds/global/type,
//   group1: data_size=2B, dims, 48-bit dim0 stride).  Issued by one wave,
//   tracked with TENSORcnt.
// ---------------------------------------------------------------------------
__device__ __forceinline__ void tdm_load_2d(unsigned lds_off, const void* gsrc,
                                            unsigned tile_x, unsigned tile_y,
                                            unsigned long long stride_elems) {
  unsigned long long ga = (unsigned long long)(uintptr_t)gsrc;
  v4u g0;
  g0[0] = 1u;                                              // count=1, no gather
  g0[1] = lds_off;                                         // lds_addr
  g0[2] = (unsigned)(ga & 0xFFFFFFFFu);                    // global_addr lo
  g0[3] = (unsigned)((ga >> 32) & 0x01FFFFFFu) | (2u << 30); // addr hi | type=2
  v8i g1;
  g1[0] = (int)(1u << 16);                                 // data_size = 2 bytes
  g1[1] = (int)((tile_x & 0xFFFFu) << 16);                 // tensor_dim0 lo16
  g1[2] = (int)((tile_x >> 16) | ((tile_y & 0xFFFFu) << 16)); // d0 hi | d1 lo
  g1[3] = (int)((tile_y >> 16) | (tile_x << 16));          // d1 hi | tile_dim0
  g1[4] = (int)(tile_y & 0xFFFFu);                         // tile_dim1, tile_dim2=0
  g1[5] = (int)(stride_elems & 0xFFFFFFFFull);             // dim0_stride lo32
  g1[6] = (int)((stride_elems >> 32) & 0xFFFFull);         // dim0_stride hi16
  g1[7] = 0;
  v4i z4 = {0, 0, 0, 0};
#if defined(__clang_major__) && (__clang_major__ >= 23)
  v8i z8 = {0, 0, 0, 0, 0, 0, 0, 0};
  __builtin_amdgcn_tensor_load_to_lds(g0, g1, z4, z4, z8, 0);
#else
  __builtin_amdgcn_tensor_load_to_lds(g0, g1, z4, z4, 0);
#endif
}

// ---------------------------------------------------------------------------
// fp32 -> bf16 conversion (weights)
// ---------------------------------------------------------------------------
__global__ __launch_bounds__(256) void cvt_bf16_kernel(const float* __restrict__ in,
                                                       u16* __restrict__ out, size_t n) {
  size_t i = (size_t)blockIdx.x * blockDim.x + threadIdx.x;
  if (i < n) out[i] = f2bf(in[i]);
}

// ---------------------------------------------------------------------------
// RMSNorm: per-row (D=2048), writes bf16
// ---------------------------------------------------------------------------
__global__ __launch_bounds__(256) void rmsnorm_kernel(const float* __restrict__ x,
                                                      const float* __restrict__ g,
                                                      u16* __restrict__ out, int D) {
  __shared__ float red[256];
  int row = blockIdx.x;
  const float* xr = x + (size_t)row * D;
  float ss = 0.f;
  for (int i = threadIdx.x; i < D; i += 256) { float v = xr[i]; ss += v * v; }
  red[threadIdx.x] = ss;
  __syncthreads();
  for (int s = 128; s > 0; s >>= 1) {
    if ((int)threadIdx.x < s) red[threadIdx.x] += red[threadIdx.x + s];
    __syncthreads();
  }
  float rinv = rsqrtf(red[0] / (float)D + 1e-6f);
  for (int i = threadIdx.x; i < D; i += 256)
    out[(size_t)row * D + i] = f2bf(xr[i] * rinv * g[i]);
}

// ---------------------------------------------------------------------------
// Tiled bf16 WMMA GEMM with TDM double-buffered LDS staging.
//   C[M,N] = A[M,K] @ B[K,N] (+bias)
//   block = 256 threads = 8 waves, block tile 64x128, K-step 32
//   wave (wm in 0..1, wn in 0..3) computes 32x32 via 2x2 v_wmma_f32_16x16x32_bf16
//   MODE 0: f32 out, no bias.  MODE 1: bf16 out + bias.  MODE 2: f32 out + bias.
// ---------------------------------------------------------------------------
#define TBM 64
#define TBN 128
#define TBK 32

template <int MODE>
__global__ __launch_bounds__(256, 2)
void gemm_bf16_kernel(const u16* __restrict__ A, const u16* __restrict__ Bm,
                      const float* __restrict__ bias,
                      float* __restrict__ Cf, u16* __restrict__ Ch,
                      int M, int N, int K) {
  __shared__ u16 As[2][TBM * TBK];   // 64x32, row-major, contiguous
  __shared__ u16 Bs[2][TBK * TBN];   // 32x128, row-major, contiguous

  const int tile_n = blockIdx.x * TBN;
  const int tile_m = blockIdx.y * TBM;
  const int tid   = threadIdx.x;
  const int wid   = tid >> 5;
  const int lane  = tid & 31;
  const int wm    = wid >> 2;     // 0..1  -> 32-row strip
  const int wn    = wid & 3;      // 0..3  -> 32-col strip
  const int khalf = lane >> 4;    // 0 / 1
  const int l16   = lane & 15;

  v8f acc[2][2];
#pragma unroll
  for (int i = 0; i < 2; ++i)
#pragma unroll
    for (int j = 0; j < 2; ++j)
#pragma unroll
      for (int r = 0; r < 8; ++r) acc[i][j][r] = 0.f;

  // prologue: TDM-stage the first K tile into buffer 0 (wave 0 issues)
  if (wid == 0) {
    tdm_load_2d((unsigned)(uintptr_t)&As[0][0], A + (size_t)tile_m * K, TBK, TBM,
                (unsigned long long)K);
    tdm_load_2d((unsigned)(uintptr_t)&Bs[0][0], Bm + tile_n, TBN, TBK,
                (unsigned long long)N);
    __builtin_amdgcn_s_wait_tensorcnt(0);
  }
  __syncthreads();

  int buf = 0;
  for (int k0 = 0; k0 < K; k0 += TBK) {
    // kick off next tile's DMA into the other buffer while we compute
    const int nb = buf ^ 1;
    if (k0 + TBK < K && wid == 0) {
      tdm_load_2d((unsigned)(uintptr_t)&As[nb][0],
                  A + (size_t)tile_m * K + (k0 + TBK), TBK, TBM,
                  (unsigned long long)K);
      tdm_load_2d((unsigned)(uintptr_t)&Bs[nb][0],
                  Bm + (size_t)(k0 + TBK) * N + tile_n, TBN, TBK,
                  (unsigned long long)N);
    }

    // Build fragments per CDNA5 wave32 layout (ISA 7.12.2)
    BF16Frag af[2], bf[2];
#pragma unroll
    for (int i = 0; i < 2; ++i) {
      const int m = wm * 32 + i * 16 + l16;
      // A frag rows are two contiguous 8-element runs: K = 8*khalf.., 16+8*khalf..
      af[i].q[0] = *reinterpret_cast<const uint4*>(&As[buf][m * TBK + 8 * khalf]);
      af[i].q[1] = *reinterpret_cast<const uint4*>(&As[buf][m * TBK + 16 + 8 * khalf]);
    }
#pragma unroll
    for (int j = 0; j < 2; ++j) {
      const int n = wn * 32 + j * 16 + l16;
#pragma unroll
      for (int e = 0; e < 16; ++e) {
        const int kk = e + 8 * khalf + 8 * (e >> 3);    // B: K index
        bf[j].u[e] = Bs[buf][kk * TBN + n];
      }
    }
#pragma unroll
    for (int i = 0; i < 2; ++i)
#pragma unroll
      for (int j = 0; j < 2; ++j)
        acc[i][j] = __builtin_amdgcn_wmma_f32_16x16x32_bf16(
            false, af[i].v, false, bf[j].v, (short)0, acc[i][j], false, false);

    // wait for the in-flight DMA, then swap buffers
    if (wid == 0) __builtin_amdgcn_s_wait_tensorcnt(0);
    __syncthreads();
    buf = nb;
  }

  // epilogue: C/D layout M = r + 8*khalf, N = lane&15
#pragma unroll
  for (int i = 0; i < 2; ++i)
#pragma unroll
    for (int j = 0; j < 2; ++j) {
      const int col = tile_n + wn * 32 + j * 16 + l16;
      const float bia = (MODE >= 1) ? bias[col] : 0.f;
#pragma unroll
      for (int r = 0; r < 8; ++r) {
        const int row = tile_m + wm * 32 + i * 16 + r + 8 * khalf;
        float v = acc[i][j][r] + bia;
        if (MODE == 1) Ch[(size_t)row * N + col] = f2bf(v);
        else           Cf[(size_t)row * N + col] = v;
      }
    }
}

// ---------------------------------------------------------------------------
// Flash attention (causal), one wave per 16-row Q tile, D_K = 128.
//   scores: 4x v_wmma (K-dim = d), PV: 8x v_wmma per 32-wide KV step.
//   Q/K/V/O laid out [b, s, h*128 + d] (GEMM output layout, no transpose).
// ---------------------------------------------------------------------------
__global__ __launch_bounds__(256)
void flash_attn_kernel(const u16* __restrict__ Qb, const u16* __restrict__ Kb,
                       const u16* __restrict__ Vb, u16* __restrict__ Ob) {
  const int S = 2048, D = 2048;
  const float scale = 0.08838834764831845f;  // 1/sqrt(128)

  const int wid   = threadIdx.x >> 5;
  const int lane  = threadIdx.x & 31;
  const int khalf = lane >> 4;
  const int l16   = lane & 15;

  const int qt = blockIdx.x * 8 + wid;   // q tile index
  const int h  = blockIdx.y;
  const int b  = blockIdx.z;
  const int q0 = qt * 16;
  const size_t base = (size_t)b * S * D + (size_t)h * 128;

  __shared__ u16 Pst[8][16 * 32];        // per-wave P staging

  // Q A-fragments for d-chunks 0..3
  BF16Frag aq[4];
#pragma unroll
  for (int t = 0; t < 4; ++t)
#pragma unroll
    for (int e = 0; e < 16; ++e) {
      const int kk = e + 8 * ((e >> 3) + khalf);
      aq[t].u[e] = Qb[base + (size_t)(q0 + l16) * D + t * 32 + kk];
    }

  float m_i[8], l_i[8];
  v8f acc[8];
#pragma unroll
  for (int r = 0; r < 8; ++r) { m_i[r] = -3.0e38f; l_i[r] = 0.f; }
#pragma unroll
  for (int t = 0; t < 8; ++t)
#pragma unroll
    for (int r = 0; r < 8; ++r) acc[t][r] = 0.f;

  for (int kv0 = 0; kv0 < q0 + 16; kv0 += 32) {
    // ---- scores S = Q K^T for two 16-col halves ----
    v8f s[2];
#pragma unroll
    for (int j = 0; j < 2; ++j) {
#pragma unroll
      for (int r = 0; r < 8; ++r) s[j][r] = 0.f;
#pragma unroll
      for (int t = 0; t < 4; ++t) {
        BF16Frag bk;  // B[d][kv] = K[kv][d]
#pragma unroll
        for (int e = 0; e < 16; ++e)
          bk.u[e] = Kb[base + (size_t)(kv0 + j * 16 + l16) * D + t * 32 + e + 8 * khalf + 8 * (e >> 3)];
        s[j] = __builtin_amdgcn_wmma_f32_16x16x32_bf16(
            false, aq[t].v, false, bk.v, (short)0, s[j], false, false);
      }
    }

    // ---- causal mask + online softmax (rows split across 16-lane groups) ----
#pragma unroll
    for (int r = 0; r < 8; ++r) {
      const int row = q0 + r + 8 * khalf;
#pragma unroll
      for (int j = 0; j < 2; ++j) {
        const int col = kv0 + j * 16 + l16;
        const float v = s[j][r] * scale;
        s[j][r] = (col <= row) ? v : -3.0e38f;
      }
      float tmax = fmaxf(s[0][r], s[1][r]);
      for (int off = 8; off >= 1; off >>= 1)
        tmax = fmaxf(tmax, __shfl_xor(tmax, off, 16));
      const float mnew = fmaxf(m_i[r], tmax);
      const float fac  = __expf(m_i[r] - mnew);
      const float p0   = __expf(s[0][r] - mnew);
      const float p1   = __expf(s[1][r] - mnew);
      s[0][r] = p0; s[1][r] = p1;
      float rs = p0 + p1;
      for (int off = 8; off >= 1; off >>= 1)
        rs += __shfl_xor(rs, off, 16);
      l_i[r] = l_i[r] * fac + rs;
      m_i[r] = mnew;
#pragma unroll
      for (int t = 0; t < 8; ++t) acc[t][r] *= fac;
    }

    // ---- P through LDS (C-layout -> A-layout); DS ops are in-order per wave ----
#pragma unroll
    for (int r = 0; r < 8; ++r) {
      const int rr = r + 8 * khalf;
      Pst[wid][rr * 32 + l16]      = f2bf(s[0][r]);
      Pst[wid][rr * 32 + 16 + l16] = f2bf(s[1][r]);
    }
    __builtin_amdgcn_wave_barrier();

    BF16Frag pa;
    pa.q[0] = *reinterpret_cast<const uint4*>(&Pst[wid][l16 * 32 + 8 * khalf]);
    pa.q[1] = *reinterpret_cast<const uint4*>(&Pst[wid][l16 * 32 + 16 + 8 * khalf]);

    // ---- O += P V ----
#pragma unroll
    for (int t = 0; t < 8; ++t) {
      BF16Frag bv;  // B[kv][d]
#pragma unroll
      for (int e = 0; e < 16; ++e)
        bv.u[e] = Vb[base + (size_t)(kv0 + e + 8 * khalf + 8 * (e >> 3)) * D + t * 16 + l16];
      acc[t] = __builtin_amdgcn_wmma_f32_16x16x32_bf16(
          false, pa.v, false, bv.v, (short)0, acc[t], false, false);
    }
  }

  // ---- normalize + store bf16 ----
#pragma unroll
  for (int t = 0; t < 8; ++t)
#pragma unroll
    for (int r = 0; r < 8; ++r) {
      const int row = q0 + r + 8 * khalf;
      const float v = acc[t][r] / l_i[r];
      Ob[base + (size_t)row * D + t * 16 + l16] = f2bf(v);
    }
}

// ---------------------------------------------------------------------------
// elementwise kernels
// ---------------------------------------------------------------------------
__global__ __launch_bounds__(256) void add_kernel(const float* __restrict__ a,
                                                  const float* __restrict__ b,
                                                  float* __restrict__ o, size_t n) {
  size_t i = (size_t)blockIdx.x * blockDim.x + threadIdx.x;
  if (i < n) o[i] = a[i] + b[i];
}

__global__ __launch_bounds__(256) void silu_mul_kernel(const float* __restrict__ a,
                                                       const float* __restrict__ b3,
                                                       u16* __restrict__ o, size_t n) {
  size_t i = (size_t)blockIdx.x * blockDim.x + threadIdx.x;
  if (i < n) {
    float x = a[i];
    float s = x / (1.f + __expf(-x));
    o[i] = f2bf(s * b3[i]);
  }
}

// ---------------------------------------------------------------------------
// host launcher
// ---------------------------------------------------------------------------
extern "C" void kernel_launch(void* const* d_in, const int* in_sizes, int n_in,
                              void* d_out, int out_size, void* d_ws, size_t ws_size,
                              hipStream_t stream) {
  (void)in_sizes; (void)n_in; (void)out_size; (void)ws_size;

  const int Dm = 2048, F = 5504, Bt = 2, St = 2048, H = 16;
  const int Mr = Bt * St;  // 4096 rows

  const float* x  = (const float*)d_in[0];
  // d_in[1] = mask (causality is computed analytically)
  const float* wq = (const float*)d_in[2];
  const float* bq = (const float*)d_in[3];
  const float* wk = (const float*)d_in[4];
  const float* bk = (const float*)d_in[5];
  const float* wv = (const float*)d_in[6];
  const float* bv = (const float*)d_in[7];
  const float* wo = (const float*)d_in[8];
  const float* bo = (const float*)d_in[9];
  const float* g1 = (const float*)d_in[10];
  const float* g2 = (const float*)d_in[11];
  const float* w1 = (const float*)d_in[12];
  const float* w3 = (const float*)d_in[13];
  const float* w2 = (const float*)d_in[14];
  float* out = (float*)d_out;

  // workspace carve (256B aligned)
  char* wsp = (char*)d_ws;
  auto take = [&](size_t bytes) -> void* {
    void* p = (void*)wsp;
    wsp += (bytes + 255) & ~(size_t)255;
    return p;
  };
  u16* wq_bf = (u16*)take((size_t)Dm * Dm * 2);
  u16* wk_bf = (u16*)take((size_t)Dm * Dm * 2);
  u16* wv_bf = (u16*)take((size_t)Dm * Dm * 2);
  u16* wo_bf = (u16*)take((size_t)Dm * Dm * 2);
  u16* w1_bf = (u16*)take((size_t)Dm * F * 2);
  u16* w3_bf = (u16*)take((size_t)Dm * F * 2);
  u16* w2_bf = (u16*)take((size_t)F * Dm * 2);
  u16* n_bf  = (u16*)take((size_t)Mr * Dm * 2);
  u16* q_bf  = (u16*)take((size_t)Mr * Dm * 2);
  u16* k_bf  = (u16*)take((size_t)Mr * Dm * 2);
  u16* v_bf  = (u16*)take((size_t)Mr * Dm * 2);
  u16* ao_bf = (u16*)take((size_t)Mr * Dm * 2);
  float* x1  = (float*)take((size_t)Mr * Dm * 4);
  float* tmp = (float*)take((size_t)Mr * Dm * 4);
  float* h1  = (float*)take((size_t)Mr * F * 4);
  float* h3  = (float*)take((size_t)Mr * F * 4);
  u16* hb_bf = (u16*)take((size_t)Mr * F * 2);

  auto cvt = [&](const float* src, u16* dst, size_t n) {
    cvt_bf16_kernel<<<dim3((unsigned)((n + 255) / 256)), 256, 0, stream>>>(src, dst, n);
  };

  // weights -> bf16
  cvt(wq, wq_bf, (size_t)Dm * Dm);
  cvt(wk, wk_bf, (size_t)Dm * Dm);
  cvt(wv, wv_bf, (size_t)Dm * Dm);
  cvt(wo, wo_bf, (size_t)Dm * Dm);
  cvt(w1, w1_bf, (size_t)Dm * F);
  cvt(w3, w3_bf, (size_t)Dm * F);
  cvt(w2, w2_bf, (size_t)F * Dm);

  // n1 = rmsnorm(x, g1)
  rmsnorm_kernel<<<Mr, 256, 0, stream>>>(x, g1, n_bf, Dm);

  // QKV projections (bias + bf16 out)
  dim3 gqkv(Dm / TBN, Mr / TBM);
  gemm_bf16_kernel<1><<<gqkv, 256, 0, stream>>>(n_bf, wq_bf, bq, nullptr, q_bf, Mr, Dm, Dm);
  gemm_bf16_kernel<1><<<gqkv, 256, 0, stream>>>(n_bf, wk_bf, bk, nullptr, k_bf, Mr, Dm, Dm);
  gemm_bf16_kernel<1><<<gqkv, 256, 0, stream>>>(n_bf, wv_bf, bv, nullptr, v_bf, Mr, Dm, Dm);

  // causal flash attention
  flash_attn_kernel<<<dim3(St / 128, H, Bt), 256, 0, stream>>>(q_bf, k_bf, v_bf, ao_bf);

  // out projection (bias + f32), residual
  gemm_bf16_kernel<2><<<gqkv, 256, 0, stream>>>(ao_bf, wo_bf, bo, tmp, nullptr, Mr, Dm, Dm);
  add_kernel<<<dim3((unsigned)(((size_t)Mr * Dm + 255) / 256)), 256, 0, stream>>>(
      x, tmp, x1, (size_t)Mr * Dm);

  // n2 = rmsnorm(x1, g2)
  rmsnorm_kernel<<<Mr, 256, 0, stream>>>(x1, g2, n_bf, Dm);

  // FFN
  dim3 gff(F / TBN, Mr / TBM);
  gemm_bf16_kernel<0><<<gff, 256, 0, stream>>>(n_bf, w1_bf, nullptr, h1, nullptr, Mr, F, Dm);
  gemm_bf16_kernel<0><<<gff, 256, 0, stream>>>(n_bf, w3_bf, nullptr, h3, nullptr, Mr, F, Dm);
  silu_mul_kernel<<<dim3((unsigned)(((size_t)Mr * F + 255) / 256)), 256, 0, stream>>>(
      h1, h3, hb_bf, (size_t)Mr * F);
  gemm_bf16_kernel<0><<<gqkv, 256, 0, stream>>>(hb_bf, w2_bf, nullptr, tmp, nullptr, Mr, Dm, F);

  // final residual -> d_out
  add_kernel<<<dim3((unsigned)(((size_t)Mr * Dm + 255) / 256)), 256, 0, stream>>>(
      x1, tmp, out, (size_t)Mr * Dm);
}
